// EGNN_79551384257048
// MI455X (gfx1250) — matrix operand order repaired and checked by reading, not compiled, not run on hardware
//
#include <hip/hip_runtime.h>
#include <hip/hip_bf16.h>

typedef _Float16 f16;
typedef __attribute__((ext_vector_type(16))) _Float16 v16h;
typedef __attribute__((ext_vector_type(8)))  _Float16 v8h;
typedef __attribute__((ext_vector_type(8)))  float    v8f;

#define D_  128
#define H_  128
#define ED_ 16

__device__ __forceinline__ v8f zero_v8f() {
  v8f z = {0.f, 0.f, 0.f, 0.f, 0.f, 0.f, 0.f, 0.f};
  return z;
}

__device__ __forceinline__ v8f wmma_f16(v16h a, v16h b, v8f c) {
  // D = A(16x32 f16) * B(32x16 f16) + C(16x16 f32)
  return __builtin_amdgcn_wmma_f32_16x16x32_f16(false, a, false, b, (short)0, c,
                                                false, false);
}

// silu via native v_rcp_f32 (avoids IEEE div fixup chains)
__device__ __forceinline__ float silu_f(float v) {
  return v * __builtin_amdgcn_rcpf(1.0f + __expf(-v));
}

// A matrix 16xK row-major f16 in LDS -> v16h in contiguous k order (same order as packed B).
// lanes 0-15: row = lane, K = {k0..k0+15}
// lanes 16-31: row = lane-16, K = {k0+16..k0+31}
// ldm multiple of 8 and k0 multiple of 32 -> 16B-aligned vector loads.
__device__ __forceinline__ v16h load_A_lds(const f16* smem, int ldm, int k0) {
  int lane = threadIdx.x & 31;
  int row  = lane & 15;
  int kh   = (lane & 16) ? 16 : 0;
  const f16* p = smem + row * ldm + k0 + kh;
  v8h lo = *(const v8h*)(p);
  v8h hi = *(const v8h*)(p + 8);
  return __builtin_shufflevector(lo, hi, 0, 1, 2, 3, 4, 5, 6, 7, 8, 9, 10, 11,
                                 12, 13, 14, 15);
}

// B fragment from pre-packed weights: tile (kt, ct) of a Kp x N matrix packed
// as 512 halfs per 32x16 tile, 16 contiguous halfs per lane.
__device__ __forceinline__ v16h load_B_packed(const f16* P, int nct, int kt,
                                              int ct) {
  int lane = threadIdx.x & 31;
  const f16* p = P + ((size_t)(kt * nct + ct) << 9) + (lane << 4);
  return *(const v16h*)p;
}

// ---------------------------------------------------------------------------
// Pack W [L, K, N] (f32 row-major) into WMMA B-fragment tiles (f16), K padded
// to Kp (multiple of 32). Tile (kt, ct): lane l holds col = ct*16 + (l&15),
// k = kt*32 + ((l&16)?16:0) + i for i in 0..15, contiguous per lane.
// ---------------------------------------------------------------------------
__global__ void egnn_pack_B(const float* __restrict__ W, f16* __restrict__ P,
                            int K, int Nc, int Kp, int total) {
  int idx = blockIdx.x * blockDim.x + threadIdx.x;
  if (idx >= total) return;
  int halfsPerMat = (Kp >> 5) * (Nc >> 4) * 512;
  int l    = idx / halfsPerMat;
  int r    = idx - l * halfsPerMat;
  int tile = r >> 9;
  int t2   = r & 511;
  int lane = t2 >> 4;
  int i    = t2 & 15;
  int nct  = Nc >> 4;
  int kt   = tile / nct;
  int ct   = tile - kt * nct;
  int col  = ct * 16 + (lane & 15);
  int k    = kt * 32 + ((lane & 16) ? 16 : 0) + i;
  float v  = (k < K) ? W[(size_t)l * K * Nc + (size_t)k * Nc + col] : 0.0f;
  P[idx]   = (f16)v;
}

// ---------------------------------------------------------------------------
// Edge pipeline: edge MLP -> node message MLP -> atomic segment-sum into agg
// One workgroup (8 waves) handles 16 edges; wave w owns output cols [16w,16w+16)
// ---------------------------------------------------------------------------
__global__ __launch_bounds__(256)
void egnn_edge_kernel(const float* __restrict__ x,         // [N,128]
                      const float* __restrict__ pos,       // [N,3]
                      const int*   __restrict__ eidx,      // [2,E]
                      const float* __restrict__ eattr,     // [E,16]
                      const float* __restrict__ stats,     // invstd@14..16
                      const f16*  __restrict__ ew1, const float* __restrict__ eb1,
                      const f16*  __restrict__ ew2, const float* __restrict__ eb2,
                      const f16*  __restrict__ nw1, const float* __restrict__ nb1,
                      const f16*  __restrict__ nw2, const float* __restrict__ nb2,
                      float* __restrict__ agg,             // [N,128]
                      int N, int E) {
  __shared__ int   s_src[16];
  __shared__ int   s_dst[16];
  __shared__ float s_dist[16];
  __shared__ __align__(16) f16 A1[16][32];   // [edge_attr | dist | pad] (K=32)
  __shared__ __align__(16) f16 A3[16][384];  // [x_dst | x_src | ef]     (K=384)
  __shared__ __align__(16) f16 Hs[16][128];  // hidden activations (reused)

  (void)N;
  const int t     = threadIdx.x;
  const int wave  = t >> 5;
  const int lane  = t & 31;
  const int col0  = wave * 16;
  const int ncol  = lane & 15;
  const int mbase = (lane & 16) ? 8 : 0;
  const int e0    = blockIdx.x * 16;

  // --- edge meta: indices, rel distance under normalized positions ---------
  if (t < 16) {
    int e = e0 + t, s = 0, d = 0;
    float dist = 0.0f;
    if (e < E) {
      s = eidx[e];
      d = eidx[E + e];
      float r0 = (pos[d * 3 + 0] - pos[s * 3 + 0]) * stats[14];
      float r1 = (pos[d * 3 + 1] - pos[s * 3 + 1]) * stats[15];
      float r2 = (pos[d * 3 + 2] - pos[s * 3 + 2]) * stats[16];
      dist = sqrtf(r0 * r0 + r1 * r1 + r2 * r2);
    }
    s_src[t] = s; s_dst[t] = d; s_dist[t] = dist;
  }
  __syncthreads();

  // --- stage A1 = [edge_attr, dist] padded to K=32 -------------------------
#pragma unroll
  for (int j = 0; j < 2; ++j) {
    int idx = j * 256 + t;
    int r = idx >> 5, c = idx & 31;
    int e = e0 + r;
    float v = 0.0f;
    if (e < E) {
      if (c < ED_) v = eattr[(size_t)e * ED_ + c];
      else if (c == ED_) v = s_dist[r];
    }
    A1[r][c] = (f16)v;
  }

  // --- gather x[dst], x[src] into A3 cols 0..255 (coalesced over t) --------
#pragma unroll 4
  for (int r = 0; r < 16; ++r) {
    int e = e0 + r;
    float v = 0.0f;
    if (e < E) {
      v = (t < 128) ? x[(size_t)s_dst[r] * D_ + t]
                    : x[(size_t)s_src[r] * D_ + (t - 128)];
    }
    A3[r][t] = (f16)v;
  }
  __syncthreads();

  // --- edge MLP layer 1: silu([E,17] @ [17,128] + b) -----------------------
  {
    v8f acc = zero_v8f();
    v16h a = load_A_lds(&A1[0][0], 32, 0);
    v16h b = load_B_packed(ew1, 8, 0, wave);
    acc = wmma_f16(a, b, acc);
    float bias = eb1[col0 + ncol];
#pragma unroll
    for (int i = 0; i < 8; ++i)
      Hs[mbase + i][col0 + ncol] = (f16)silu_f(acc[i] + bias);
  }
  __syncthreads();

  // --- edge MLP layer 2: ef = H @ [128,128] + b -> A3 cols 256..383 --------
  {
    v8f acc = zero_v8f();
#pragma unroll
    for (int kt = 0; kt < 4; ++kt) {
      v16h a = load_A_lds(&Hs[0][0], H_, kt * 32);
      v16h b = load_B_packed(ew2, 8, kt, wave);
      acc = wmma_f16(a, b, acc);
    }
    float bias = eb2[col0 + ncol];
#pragma unroll
    for (int i = 0; i < 8; ++i)
      A3[mbase + i][256 + col0 + ncol] = (f16)(acc[i] + bias);
  }
  __syncthreads();

  // --- node message MLP layer 1: silu([E,384] @ [384,128] + b) -------------
  {
    v8f acc = zero_v8f();
#pragma unroll
    for (int kt = 0; kt < 12; ++kt) {
      v16h a = load_A_lds(&A3[0][0], 384, kt * 32);
      v16h b = load_B_packed(nw1, 8, kt, wave);
      acc = wmma_f16(a, b, acc);
    }
    float bias = nb1[col0 + ncol];
#pragma unroll
    for (int i = 0; i < 8; ++i)
      Hs[mbase + i][col0 + ncol] = (f16)silu_f(acc[i] + bias);
  }
  __syncthreads();

  // --- node message MLP layer 2 + mean-aggregation (segment sum) -----------
  {
    v8f acc = zero_v8f();
#pragma unroll
    for (int kt = 0; kt < 4; ++kt) {
      v16h a = load_A_lds(&Hs[0][0], H_, kt * 32);
      v16h b = load_B_packed(nw2, 8, kt, wave);
      acc = wmma_f16(a, b, acc);
    }
    float bias = nb2[col0 + ncol];
#pragma unroll
    for (int i = 0; i < 8; ++i) {
      int r = mbase + i;
      int e = e0 + r;
      if (e < E)
        atomicAdd(&agg[(size_t)s_dst[r] * D_ + col0 + ncol], acc[i] + bias);
    }
  }
}

// ---------------------------------------------------------------------------
// Node pipeline: mean-agg + residual + LayerNorm -> x_next; coord MLP -> coef
// ---------------------------------------------------------------------------
__global__ __launch_bounds__(256)
void egnn_node_kernel(const float* __restrict__ x,
                      const float* __restrict__ agg,
                      const float* __restrict__ counts,
                      const float* __restrict__ ln_g,
                      const float* __restrict__ ln_b,
                      const f16*  __restrict__ cw1, const float* __restrict__ cb1,
                      const float* __restrict__ cw2,   // [128] f32
                      float* __restrict__ x_next,
                      float* __restrict__ coef,
                      int N) {
  __shared__ float O[16][128];
  __shared__ __align__(16) f16 Af[16][128];
  __shared__ __align__(16) f16 Hs[16][128];
  __shared__ float s_mu[16];
  __shared__ float s_rs[16];

  const int t     = threadIdx.x;
  const int wave  = t >> 5;
  const int lane  = t & 31;
  const int col0  = wave * 16;
  const int ncol  = lane & 15;
  const int mbase = (lane & 16) ? 8 : 0;
  const int n0    = blockIdx.x * 16;

  // out0 = agg/counts + x  (native rcp for the mean)
#pragma unroll
  for (int r = 0; r < 16; r += 2) {
    int rr = r + (t >> 7);
    int c  = t & 127;
    int n  = n0 + rr;
    float v = 0.0f;
    if (n < N) {
      float rcnt = __builtin_amdgcn_rcpf(fmaxf(counts[n], 1.0f));
      v = agg[(size_t)n * D_ + c] * rcnt + x[(size_t)n * D_ + c];
    }
    O[rr][c] = v;
  }
  __syncthreads();

  // LayerNorm statistics (per node)
  if (t < 16) {
    float mu = 0.0f;
#pragma unroll 8
    for (int c = 0; c < 128; ++c) mu += O[t][c];
    mu *= (1.0f / 128.0f);
    float var = 0.0f;
#pragma unroll 8
    for (int c = 0; c < 128; ++c) { float d = O[t][c] - mu; var += d * d; }
    var *= (1.0f / 128.0f);
    s_mu[t] = mu;
    s_rs[t] = rsqrtf(var + 1e-5f);
  }
  __syncthreads();

  // normalize -> x_next, f16 copy for WMMA
#pragma unroll
  for (int r = 0; r < 16; r += 2) {
    int rr = r + (t >> 7);
    int c  = t & 127;
    int n  = n0 + rr;
    float o = (O[rr][c] - s_mu[rr]) * s_rs[rr] * ln_g[c] + ln_b[c];
    if (n < N) x_next[(size_t)n * D_ + c] = o;
    Af[rr][c] = (f16)o;
  }
  __syncthreads();

  // coord MLP hidden: silu(out @ cw1 + cb1)
  {
    v8f acc = zero_v8f();
#pragma unroll
    for (int kt = 0; kt < 4; ++kt) {
      v16h a = load_A_lds(&Af[0][0], H_, kt * 32);
      v16h b = load_B_packed(cw1, 8, kt, wave);
      acc = wmma_f16(a, b, acc);
    }
    float bias = cb1[col0 + ncol];
#pragma unroll
    for (int i = 0; i < 8; ++i)
      Hs[mbase + i][col0 + ncol] = (f16)silu_f(acc[i] + bias);
  }
  __syncthreads();

  // coef = hidden @ cw2 ([128,1])
  if (t < 16) {
    float s = 0.0f;
#pragma unroll 8
    for (int c = 0; c < 128; ++c) s += (float)Hs[t][c] * cw2[c];
    int n = n0 + t;
    if (n < N) coef[n] = s;
  }
}

// ---------------------------------------------------------------------------
// Small VALU kernels
// ---------------------------------------------------------------------------
__global__ void egnn_pos_stats(const float* __restrict__ pos,
                               float* __restrict__ stats, int N) {
  __shared__ float sh[6];
  if (threadIdx.x < 6) sh[threadIdx.x] = 0.0f;
  __syncthreads();
  float s[3] = {0, 0, 0}, q[3] = {0, 0, 0};
  for (int n = blockIdx.x * blockDim.x + threadIdx.x; n < N;
       n += gridDim.x * blockDim.x) {
#pragma unroll
    for (int k = 0; k < 3; ++k) {
      float v = pos[n * 3 + k];
      s[k] += v;
      q[k] += v * v;
    }
  }
#pragma unroll
  for (int k = 0; k < 3; ++k) {
    atomicAdd(&sh[k], s[k]);
    atomicAdd(&sh[3 + k], q[k]);
  }
  __syncthreads();
  if (threadIdx.x < 6) atomicAdd(&stats[threadIdx.x], sh[threadIdx.x]);
}

__global__ void egnn_stats_fin(float* __restrict__ stats, int N) {
  if (blockIdx.x == 0 && threadIdx.x == 0) {
#pragma unroll
    for (int k = 0; k < 3; ++k) {
      float mean = stats[k] / (float)N;
      float var  = (stats[3 + k] - (float)N * mean * mean) / (float)(N - 1);
      float sd   = sqrtf(fmaxf(var, 0.0f));
      if (sd < 1e-8f) sd = 1e-8f;
      stats[8 + k]  = mean;
      stats[11 + k] = sd;
      stats[14 + k] = 1.0f / sd;
    }
  }
}

__global__ void egnn_counts(const int* __restrict__ eidx,
                            float* __restrict__ counts, int E) {
  int e = blockIdx.x * blockDim.x + threadIdx.x;
  if (e < E) atomicAdd(&counts[eidx[E + e]], 1.0f);
}

__global__ void egnn_pos_edge(const int* __restrict__ eidx,
                              const float* __restrict__ pos,
                              const float* __restrict__ stats,
                              const float* __restrict__ coef,
                              float* __restrict__ pacc, int E) {
  int e = blockIdx.x * blockDim.x + threadIdx.x;
  if (e >= E) return;
  int s = eidx[e], d = eidx[E + e];
  float c = coef[d];
#pragma unroll
  for (int k = 0; k < 3; ++k) {
    float rel = (pos[d * 3 + k] - pos[s * 3 + k]) * stats[14 + k];
    atomicAdd(&pacc[d * 3 + k], c * rel);
  }
}

__global__ void egnn_pos_apply(float* __restrict__ pos,
                               const float* __restrict__ pacc,
                               const float* __restrict__ stats, int N3) {
  int i = blockIdx.x * blockDim.x + threadIdx.x;
  if (i < N3) pos[i] += pacc[i] * stats[11 + (i % 3)];
}

// ---------------------------------------------------------------------------
extern "C" void kernel_launch(void* const* d_in, const int* in_sizes, int n_in,
                              void* d_out, int out_size, void* d_ws,
                              size_t ws_size, hipStream_t stream) {
  (void)n_in; (void)out_size; (void)ws_size;
  const float* x_in   = (const float*)d_in[0];
  const float* pos_in = (const float*)d_in[1];
  const int*   eidx   = (const int*)d_in[2];
  const float* eattr  = (const float*)d_in[3];
  const float* ew1f   = (const float*)d_in[4];
  const float* eb1    = (const float*)d_in[5];
  const float* ew2f   = (const float*)d_in[6];
  const float* eb2    = (const float*)d_in[7];
  const float* nw1f   = (const float*)d_in[8];
  const float* nb1    = (const float*)d_in[9];
  const float* nw2f   = (const float*)d_in[10];
  const float* nb2    = (const float*)d_in[11];
  const float* cw1f   = (const float*)d_in[12];
  const float* cb1    = (const float*)d_in[13];
  const float* cw2    = (const float*)d_in[14];
  const float* lng    = (const float*)d_in[15];
  const float* lnb    = (const float*)d_in[16];

  const int N = in_sizes[0] / D_;
  const int E = in_sizes[2] / 2;
  const int L = in_sizes[5] / H_;

  // Packed B sizes (halfs) per layer: tiles(Kp/32 * N/16) * 512
  const size_t ew1_ph = 1  * 8 * 512;   // Kp=32,  N=128
  const size_t ew2_ph = 4  * 8 * 512;   // Kp=128, N=128
  const size_t nw1_ph = 12 * 8 * 512;   // Kp=384, N=128
  const size_t nw2_ph = 4  * 8 * 512;   // Kp=128, N=128
  const size_t cw1_ph = 4  * 8 * 512;   // Kp=128, N=128

  // --- workspace carve-out (256B aligned) ---------------------------------
  char* p = (char*)d_ws;
  auto carve = [&](size_t bytes) {
    char* r = p;
    p += (bytes + 255) & ~(size_t)255;
    return r;
  };
  float* xa     = (float*)carve((size_t)N * D_ * 4);
  float* xb     = (float*)carve((size_t)N * D_ * 4);
  float* agg    = (float*)carve((size_t)N * D_ * 4);
  float* counts = (float*)carve((size_t)N * 4);
  float* coef   = (float*)carve((size_t)N * 4);
  float* posb   = (float*)carve((size_t)N * 3 * 4);
  float* pacc   = (float*)carve((size_t)N * 3 * 4);
  float* stats  = (float*)carve(64 * 4);
  f16* ew1h = (f16*)carve((size_t)L * ew1_ph * 2);
  f16* ew2h = (f16*)carve((size_t)L * ew2_ph * 2);
  f16* nw1h = (f16*)carve((size_t)L * nw1_ph * 2);
  f16* nw2h = (f16*)carve((size_t)L * nw2_ph * 2);
  f16* cw1h = (f16*)carve((size_t)L * cw1_ph * 2);

  // --- weight packing into WMMA B-fragment layout -------------------------
  auto pack = [&](const float* W, f16* P, int K, int Nc, int Kp) {
    int total = L * (Kp >> 5) * (Nc >> 4) * 512;
    egnn_pack_B<<<(total + 255) / 256, 256, 0, stream>>>(W, P, K, Nc, Kp,
                                                         total);
  };
  pack(ew1f, ew1h, ED_ + 1,    H_, 32);
  pack(ew2f, ew2h, H_,         H_, 128);
  pack(nw1f, nw1h, 2 * D_ + H_, H_, 384);
  pack(nw2f, nw2h, H_,         D_, 128);
  pack(cw1f, cw1h, H_,         H_, 128);

  // --- init: x ping-pong, pos buffer, counts ------------------------------
  hipMemcpyAsync(xa, x_in, (size_t)N * D_ * 4, hipMemcpyDeviceToDevice, stream);
  hipMemcpyAsync(posb, pos_in, (size_t)N * 3 * 4, hipMemcpyDeviceToDevice,
                 stream);
  hipMemsetAsync(counts, 0, (size_t)N * 4, stream);
  egnn_counts<<<(E + 255) / 256, 256, 0, stream>>>(eidx, counts, E);

  float* xc = xa;
  float* xn = xb;
  for (int l = 0; l < L; ++l) {
    hipMemsetAsync(stats, 0, 64 * 4, stream);
    egnn_pos_stats<<<64, 256, 0, stream>>>(posb, stats, N);
    egnn_stats_fin<<<1, 64, 0, stream>>>(stats, N);

    hipMemsetAsync(agg, 0, (size_t)N * D_ * 4, stream);
    egnn_edge_kernel<<<(E + 15) / 16, 256, 0, stream>>>(
        xc, posb, eidx, eattr, stats,
        ew1h + (size_t)l * ew1_ph, eb1 + (size_t)l * H_,
        ew2h + (size_t)l * ew2_ph, eb2 + (size_t)l * H_,
        nw1h + (size_t)l * nw1_ph, nb1 + (size_t)l * H_,
        nw2h + (size_t)l * nw2_ph, nb2 + (size_t)l * D_,
        agg, N, E);

    egnn_node_kernel<<<(N + 15) / 16, 256, 0, stream>>>(
        xc, agg, counts, lng + (size_t)l * D_, lnb + (size_t)l * D_,
        cw1h + (size_t)l * cw1_ph, cb1 + (size_t)l * H_,
        cw2 + (size_t)l * H_, xn, coef, N);

    hipMemsetAsync(pacc, 0, (size_t)N * 3 * 4, stream);
    egnn_pos_edge<<<(E + 255) / 256, 256, 0, stream>>>(eidx, posb, stats, coef,
                                                       pacc, E);
    egnn_pos_apply<<<(N * 3 + 255) / 256, 256, 0, stream>>>(posb, pacc, stats,
                                                            N * 3);
    float* tmp = xc; xc = xn; xn = tmp;
  }

  hipMemcpyAsync(d_out, xc, (size_t)N * D_ * 4, hipMemcpyDeviceToDevice,
                 stream);
  hipMemcpyAsync((float*)d_out + (size_t)N * D_, posb, (size_t)N * 3 * 4,
                 hipMemcpyDeviceToDevice, stream);
}
